// FlashAttentionSimulator_21835613733532
// MI455X (gfx1250) — matrix operand order repaired and checked by reading, not compiled, not run on hardware
//
#include <hip/hip_runtime.h>
#include <stdint.h>

// Problem constants (match reference)
#define B_    2
#define S_    2048
#define DM_   1024
#define H_    16
#define DK_   64
#define ROWS_ (B_ * S_)   // 4096

typedef __attribute__((ext_vector_type(16))) __bf16 v16bf;
typedef __attribute__((ext_vector_type(8)))  float  v8f;

union ABFrag { uint4 u[2]; unsigned w[8]; v16bf v; };

__device__ __forceinline__ unsigned short f2bf(float f) {   // RNE, for epilogues
  union { float f; unsigned u; } c; c.f = f;
  unsigned r = c.u + 0x7FFFu + ((c.u >> 16) & 1u);
  return (unsigned short)(r >> 16);
}

// pack two f32 -> packed bf16 pair (round-half-up via +0x8000, then byte-perm)
__device__ __forceinline__ unsigned pack_bf16(float lo, float hi) {
  union { float f; unsigned u; } a, b;
  a.f = lo; b.f = hi;
  return __builtin_amdgcn_perm(b.u + 0x8000u, a.u + 0x8000u, 0x07060302u);
}

// ---------------------------------------------------------------------------
// fp32 -> bf16 conversion, 8 elements per thread
// ---------------------------------------------------------------------------
__global__ void cvt_f32_to_bf16(const float* __restrict__ in,
                                unsigned short* __restrict__ out, int n8) {
  int i = blockIdx.x * blockDim.x + threadIdx.x;
  if (i >= n8) return;
  const float4* p = (const float4*)in + (size_t)i * 2;
  float4 a = p[0], b = p[1];
  uint4 o;
  o.x = (unsigned)f2bf(a.x) | ((unsigned)f2bf(a.y) << 16);
  o.y = (unsigned)f2bf(a.z) | ((unsigned)f2bf(a.w) << 16);
  o.z = (unsigned)f2bf(b.x) | ((unsigned)f2bf(b.y) << 16);
  o.w = (unsigned)f2bf(b.z) | ((unsigned)f2bf(b.w) << 16);
  ((uint4*)out)[i] = o;
}

// ---------------------------------------------------------------------------
// GEMM:  Y[m][n] = (sum_k X[m][k] * W[n][k] + bias[n]) * out_scale
//   One wave computes a 32(M) x 64(N) tile: 8 WMMA accumulators, B reused x2.
//   LAYOUT: 1 = bf16 (B,H,S,DK); 2 = bf16 (B,H,DK,S); 3 = f32 row-major
// ---------------------------------------------------------------------------
template <int LAYOUT>
__global__ __launch_bounds__(128) void gemm_xwT(
    const unsigned short* __restrict__ X,
    const unsigned short* __restrict__ W,
    const float* __restrict__ bias,
    void* __restrict__ Y, float out_scale) {
  const int lane = threadIdx.x & 31;
  const int wave = threadIdx.x >> 5;
  const int lm   = lane & 15;
  const int half = lane >> 4;
  const int n0 = blockIdx.x * 64;
  const int m0 = (blockIdx.y * 4 + wave) * 32;

  v8f acc[8] = {};
  const unsigned short* ar0 = X + (size_t)(m0 + lm) * DM_;
  const unsigned short* ar1 = X + (size_t)(m0 + 16 + lm) * DM_;

  for (int k0 = 0; k0 < DM_; k0 += 32) {
    ABFrag a0, a1;  // A 16x32: half0 -> K{0..7,16..23}, half1 -> K{8..15,24..31}
    a0.u[0] = *(const uint4*)(ar0 + k0 + half * 8);
    a0.u[1] = *(const uint4*)(ar0 + k0 + half * 8 + 16);
    a1.u[0] = *(const uint4*)(ar1 + k0 + half * 8);
    a1.u[1] = *(const uint4*)(ar1 + k0 + half * 8 + 16);
#pragma unroll
    for (int t = 0; t < 4; ++t) {
      const unsigned short* wr =
          W + (size_t)(n0 + t * 16 + lm) * DM_ + k0 + half * 8;
      ABFrag b;  // B 32x16: lane = column n, K = half*8..+7, 16+half*8..+7
      b.u[0] = *(const uint4*)(wr);
      b.u[1] = *(const uint4*)(wr + 16);
      acc[t] = __builtin_amdgcn_wmma_f32_16x16x32_bf16(
          false, a0.v, false, b.v, (short)0, acc[t], false, false);
      acc[4 + t] = __builtin_amdgcn_wmma_f32_16x16x32_bf16(
          false, a1.v, false, b.v, (short)0, acc[4 + t], false, false);
    }
  }

#pragma unroll
  for (int i = 0; i < 2; ++i)
#pragma unroll
    for (int t = 0; t < 4; ++t) {
      const int n = n0 + t * 16 + lm;
      const float bv = bias[n];
#pragma unroll
      for (int r = 0; r < 8; ++r) {
        const int m = m0 + i * 16 + r + half * 8;
        const float v = (acc[i * 4 + t][r] + bv) * out_scale;
        if constexpr (LAYOUT == 1) {        // (B,H,S,DK) bf16
          const int b = m >> 11, s = m & (S_ - 1);
          const int h = n >> 6,  d = n & (DK_ - 1);
          ((unsigned short*)Y)[(((size_t)(b * H_ + h) * S_ + s) * DK_) + d] = f2bf(v);
        } else if constexpr (LAYOUT == 2) { // (B,H,DK,S) bf16 (V transposed)
          const int b = m >> 11, s = m & (S_ - 1);
          const int h = n >> 6,  d = n & (DK_ - 1);
          ((unsigned short*)Y)[(((size_t)(b * H_ + h) * DK_ + d) * S_) + s] = f2bf(v);
        } else {                            // row-major f32
          ((float*)Y)[(size_t)m * DM_ + n] = v;
        }
      }
    }
}

// ---------------------------------------------------------------------------
// Causal attention, transposed-score formulation. One wave per 16-row q-tile.
//   Q: (B,H,S,64) bf16, PRE-SCALED by (1/sqrt(64))*log2(e) in the projection.
//   K: (B,H,S,64) bf16 ; Vt: (B,H,64,S) bf16 ; AO: [4096][1024] bf16
//   S^T tile = K_tile(A) x Q^T(B)  ->  C-layout has q in lanes, kv in VGPRs,
//   which IS the A-operand layout for P x V. No LDS.
//   Softmax denominator = P x Ones via one extra WMMA: every lane's slot r
//   holds sum_kv p[q=r+8*half][kv] -> no cross-lane shuffles anywhere.
// ---------------------------------------------------------------------------
__global__ __launch_bounds__(128) void attn_kernel(
    const unsigned short* __restrict__ Q,
    const unsigned short* __restrict__ K,
    const unsigned short* __restrict__ Vt,
    unsigned short* __restrict__ AO) {
  const int lane = threadIdx.x & 31;
  const int wave = threadIdx.x >> 5;
  const int lm   = lane & 15;
  const int half = lane >> 4;
  const int bh = blockIdx.y;                    // b*H + h
  const int q0 = (blockIdx.x * 4 + wave) * 16;  // q-tile base row

  const unsigned short* Qb = Q  + (size_t)bh * S_ * DK_;
  const unsigned short* Kb = K  + (size_t)bh * S_ * DK_;
  const unsigned short* Vb = Vt + (size_t)bh * DK_ * S_;

  // Q as B-operand (Q^T, 32d x 16q per step): lane = q column, d = half*16..+15
  ABFrag qb0, qb1;
  {
    const unsigned short* qrow = Qb + (size_t)(q0 + lm) * DK_;
    qb0.u[0] = *(const uint4*)(qrow + half * 16);
    qb0.u[1] = *(const uint4*)(qrow + half * 16 + 8);
    qb1.u[0] = *(const uint4*)(qrow + 32 + half * 16);
    qb1.u[1] = *(const uint4*)(qrow + 32 + half * 16 + 8);
  }

  // constant all-ones B operand (bf16 1.0 pairs) for the denominator WMMA
  ABFrag ones;
#pragma unroll
  for (int j = 0; j < 8; ++j) ones.w[j] = 0x3F803F80u;

  v8f o[4] = {};   // 16q x 64d output accumulator (q rows = r + 8*half)
  v8f o4  = {};    // denominator accumulator: sum_kv p (same C layout)
  const int qcol = q0 + lm;

  auto chunk = [&](int kv0, bool masked) {
    // ---- S^T: two 16(kv) x 16(q) tiles ----
    v8f sc[2];
#pragma unroll
    for (int t = 0; t < 2; ++t) {
      const unsigned short* krow = Kb + (size_t)(kv0 + t * 16 + lm) * DK_;
      ABFrag ka0, ka1;  // K rows as A-operand, d = half*16..+15, same order as qb
      ka0.u[0] = *(const uint4*)(krow + half * 16);
      ka0.u[1] = *(const uint4*)(krow + half * 16 + 8);
      ka1.u[0] = *(const uint4*)(krow + 32 + half * 16);
      ka1.u[1] = *(const uint4*)(krow + 32 + half * 16 + 8);
      v8f c = {};
      c = __builtin_amdgcn_wmma_f32_16x16x32_bf16(false, ka0.v, false, qb0.v,
                                                  (short)0, c, false, false);
      c = __builtin_amdgcn_wmma_f32_16x16x32_bf16(false, ka1.v, false, qb1.v,
                                                  (short)0, c, false, false);
      sc[t] = c;
    }

    // ---- causal mask: only the diagonal chunk ----
    if (masked) {
#pragma unroll
      for (int t = 0; t < 2; ++t) {
        const int kvbase = kv0 + t * 16 + half * 8;
#pragma unroll
        for (int r = 0; r < 8; ++r)
          if (kvbase + r > qcol) sc[t][r] = -1.0e30f;
      }
    }

    // ---- p = exp2(s) (Q pre-scaled to exp2 domain) ----
#pragma unroll
    for (int t = 0; t < 2; ++t)
#pragma unroll
      for (int r = 0; r < 8; ++r)
        sc[t][r] = __builtin_amdgcn_exp2f(sc[t][r]);

    // ---- P is already in A-operand layout: pack in-lane to bf16 ----
    ABFrag pf;
#pragma unroll
    for (int j = 0; j < 4; ++j) {
      pf.w[j]     = pack_bf16(sc[0][2 * j], sc[0][2 * j + 1]);
      pf.w[4 + j] = pack_bf16(sc[1][2 * j], sc[1][2 * j + 1]);
    }

    // ---- O += P x V (4 WMMAs) ; denom += P x Ones (1 WMMA) ----
#pragma unroll
    for (int t = 0; t < 4; ++t) {
      const unsigned short* vrow =
          Vb + (size_t)(t * 16 + lm) * S_ + kv0 + half * 8;
      ABFrag b;
      b.u[0] = *(const uint4*)(vrow);
      b.u[1] = *(const uint4*)(vrow + 16);
      o[t] = __builtin_amdgcn_wmma_f32_16x16x32_bf16(
          false, pf.v, false, b.v, (short)0, o[t], false, false);
    }
    o4 = __builtin_amdgcn_wmma_f32_16x16x32_bf16(
        false, pf.v, false, ones.v, (short)0, o4, false, false);
  };

  const int kv_last = ((q0 + 15) >> 5) << 5;  // diagonal chunk base
  for (int kv0 = 0; kv0 < kv_last; kv0 += 32) chunk(kv0, false);
  chunk(kv_last, true);

  // ---- epilogue: per-lane normalize (denom lives in matching slot r) ----
  const int bb = bh >> 4, hh = bh & 15;
#pragma unroll
  for (int r = 0; r < 8; ++r) {
    const float inv = 1.0f / (o4[r] + 1e-6f);
    const int s = q0 + r + 8 * half;
    unsigned short* orow = AO + (size_t)(bb * S_ + s) * DM_ + hh * DK_;
#pragma unroll
    for (int t = 0; t < 4; ++t)
      orow[t * 16 + lm] = f2bf(o[t][r] * inv);
  }
}

// ---------------------------------------------------------------------------
// Host launch
// ---------------------------------------------------------------------------
extern "C" void kernel_launch(void* const* d_in, const int* in_sizes, int n_in,
                              void* d_out, int out_size, void* d_ws, size_t ws_size,
                              hipStream_t stream) {
  (void)in_sizes; (void)n_in; (void)out_size; (void)ws_size;
  const float* q_in = (const float*)d_in[0];
  const float* k_in = (const float*)d_in[1];
  const float* v_in = (const float*)d_in[2];
  const float* Wq = (const float*)d_in[3]; const float* bq = (const float*)d_in[4];
  const float* Wk = (const float*)d_in[5]; const float* bk = (const float*)d_in[6];
  const float* Wv = (const float*)d_in[7]; const float* bv = (const float*)d_in[8];
  const float* Wo = (const float*)d_in[9]; const float* bo = (const float*)d_in[10];

  char* ws = (char*)d_ws;
  size_t off = 0;
  auto take = [&](size_t bytes) -> char* {
    char* p = ws + off;
    off += (bytes + 255) & ~(size_t)255;
    return p;
  };
  const size_t actB = (size_t)ROWS_ * DM_ * sizeof(unsigned short);  // 8 MiB
  const size_t wB   = (size_t)DM_ * DM_ * sizeof(unsigned short);    // 2 MiB
  unsigned short* Xq  = (unsigned short*)take(actB);
  unsigned short* Xk  = (unsigned short*)take(actB);
  unsigned short* Xv  = (unsigned short*)take(actB);
  unsigned short* Wqb = (unsigned short*)take(wB);
  unsigned short* Wkb = (unsigned short*)take(wB);
  unsigned short* Wvb = (unsigned short*)take(wB);
  unsigned short* Wob = (unsigned short*)take(wB);
  unsigned short* Qb  = (unsigned short*)take(actB);   // (B,H,S,64), exp2-domain scaled
  unsigned short* Kb  = (unsigned short*)take(actB);   // (B,H,S,64)
  unsigned short* Vtb = (unsigned short*)take(actB);   // (B,H,64,S)
  unsigned short* AOb = (unsigned short*)take(actB);   // [4096][1024]

  const int actN8 = ROWS_ * DM_ / 8;
  const int wN8   = DM_ * DM_ / 8;
  cvt_f32_to_bf16<<<(actN8 + 255) / 256, 256, 0, stream>>>(q_in, Xq, actN8);
  cvt_f32_to_bf16<<<(actN8 + 255) / 256, 256, 0, stream>>>(k_in, Xk, actN8);
  cvt_f32_to_bf16<<<(actN8 + 255) / 256, 256, 0, stream>>>(v_in, Xv, actN8);
  cvt_f32_to_bf16<<<(wN8 + 255) / 256, 256, 0, stream>>>(Wq, Wqb, wN8);
  cvt_f32_to_bf16<<<(wN8 + 255) / 256, 256, 0, stream>>>(Wk, Wkb, wN8);
  cvt_f32_to_bf16<<<(wN8 + 255) / 256, 256, 0, stream>>>(Wv, Wvb, wN8);
  cvt_f32_to_bf16<<<(wN8 + 255) / 256, 256, 0, stream>>>(Wo, Wob, wN8);

  // Q carries (1/sqrt(dk)) * log2(e) so attention works directly in exp2 domain
  const float qscale = 0.125f * 1.4426950408889634f;
  const dim3 ggrid(DM_ / 64, ROWS_ / 128);   // (16, 32): 32 rows per wave
  gemm_xwT<1><<<ggrid, 128, 0, stream>>>(Xq, Wqb, bq, Qb, qscale);
  gemm_xwT<1><<<ggrid, 128, 0, stream>>>(Xk, Wkb, bk, Kb, 1.0f);
  gemm_xwT<2><<<ggrid, 128, 0, stream>>>(Xv, Wvb, bv, Vtb, 1.0f);

  attn_kernel<<<dim3(S_ / 64, B_ * H_), 128, 0, stream>>>(Qb, Kb, Vtb, AOb);

  gemm_xwT<3><<<ggrid, 128, 0, stream>>>(AOb, Wob, bo, d_out, 1.0f);
}
